// Griffin_27015344292125
// MI455X (gfx1250) — matrix operand order repaired and checked by reading, not compiled, not run on hardware
//
#include <hip/hip_runtime.h>
#include <hip/hip_bf16.h>
#include <math.h>

typedef __bf16 bf16;
typedef __attribute__((ext_vector_type(16))) __bf16 bf16x16;
typedef __attribute__((ext_vector_type(8)))  __bf16 bf16x8;
typedef __attribute__((ext_vector_type(8)))  float   f32x8;

// ---------------------------------------------------------------------------
// helpers
// ---------------------------------------------------------------------------
__device__ __forceinline__ float sigmoid_f(float x) { return 1.0f / (1.0f + __expf(-x)); }
__device__ __forceinline__ float gelu_f(float x) {
    return 0.5f * x * (1.0f + erff(x * 0.70710678118654752f));
}
// LDS aperture: flat address low 32 bits == LDS byte offset (ISA 10.2 aperture calc)
__device__ __forceinline__ uint32_t lds_off(const void* p) {
    return (uint32_t)(uintptr_t)p;
}

// ---------------------------------------------------------------------------
// f32 -> bf16 conversion (weights / activations)
// ---------------------------------------------------------------------------
__global__ __launch_bounds__(256) void f32_to_bf16_kernel(const float* __restrict__ s,
                                                          bf16* __restrict__ d, size_t n) {
    size_t i = (size_t)blockIdx.x * blockDim.x + threadIdx.x;
    size_t stride = (size_t)gridDim.x * blockDim.x;
    for (; i < n; i += stride) d[i] = (bf16)s[i];
}

// ---------------------------------------------------------------------------
// rmsnorm over last dim (L2 norm), out = x/||x|| * gamma * sqrt(D), in bf16
// ---------------------------------------------------------------------------
__global__ __launch_bounds__(256) void rmsnorm_bf16_kernel(const float* __restrict__ x,
                                                           const float* __restrict__ gamma,
                                                           bf16* __restrict__ out, int D,
                                                           float sqrtD) {
    __shared__ float red[8];
    const int row = blockIdx.x;
    const float* xr = x + (size_t)row * D;
    float ss = 0.0f;
    for (int c = threadIdx.x; c < D; c += 256) { float v = xr[c]; ss += v * v; }
#pragma unroll
    for (int s = 16; s > 0; s >>= 1) ss += __shfl_xor(ss, s, 32);
    if ((threadIdx.x & 31) == 0) red[threadIdx.x >> 5] = ss;
    __syncthreads();
    float tot = 0.0f;
#pragma unroll
    for (int i = 0; i < 8; i++) tot += red[i];
    const float sc = sqrtD * rsqrtf(tot);
    bf16* orow = out + (size_t)row * D;
    for (int c = threadIdx.x; c < D; c += 256) orow[c] = (bf16)(xr[c] * sc * gamma[c]);
}

// ---------------------------------------------------------------------------
// bf16 WMMA GEMM:  C[M,N] = A[M,K] @ W[N,K]^T  (+bias[N]) (+Res[M,N])
// WG = 256 threads = 8 waves, WG tile 128x128, wave tile 32x64 (2x4 wmma accs).
// A/B tiles staged in LDS via global_load_async_to_lds_b128 (ASYNCcnt),
// double-buffered, 80B row stride (16B pad) to spread LDS banks.
// Requires M%128==0, N%128==0, K%32==0.
// ---------------------------------------------------------------------------
#define LDS_STRIDE 80                       // bytes per 32-bf16 row (+16B pad)
#define TILE_BYTES (128 * LDS_STRIDE)       // 10240 B per operand tile
#define BUF_BYTES  (2 * TILE_BYTES)         // A + B     = 20480 B
__global__ __launch_bounds__(256) void gemm_bf16_wmma_kernel(
    const bf16* __restrict__ A, const bf16* __restrict__ W,
    const float* __restrict__ bias, const float* __restrict__ Res,
    float* __restrict__ C, int M, int N, int K) {
    __shared__ __align__(16) char smem[2 * BUF_BYTES];  // 40 KB double buffer

    const int tid = threadIdx.x;
    const int lane = tid & 31;
    const int wave = tid >> 5;
    const int wm = wave & 3;    // 4 waves along M (32 rows each)
    const int wn = wave >> 2;   // 2 waves along N (64 cols each)
    const int lh = lane >> 4;   // lane half
    const int l16 = lane & 15;
    const int bM = blockIdx.y * 128;
    const int bN = blockIdx.x * 128;

    // loader mapping: thread fills 32 contiguous bytes (2x b128) of one tile row
    const int lrow = tid >> 1;                 // 0..127
    const int lcol = (tid & 1) * 16;           // element offset 0 / 16
    const bf16* gA = A + (size_t)(bM + lrow) * K + lcol;
    const bf16* gB = W + (size_t)(bN + lrow) * K + lcol;
    const uint32_t sbase = lds_off(smem);
    const uint32_t laA = sbase + lrow * LDS_STRIDE + (tid & 1) * 32;
    const uint32_t laB = laA + TILE_BYTES;

    auto issue = [&](int buf, int k0) {
        const uint32_t bo = (uint32_t)buf * BUF_BYTES;
        uint64_t a0 = (uint64_t)(uintptr_t)(gA + k0);
        uint64_t b0 = (uint64_t)(uintptr_t)(gB + k0);
        asm volatile("global_load_async_to_lds_b128 %0, %1, off"
                     :: "v"(laA + bo), "v"(a0) : "memory");
        asm volatile("global_load_async_to_lds_b128 %0, %1, off"
                     :: "v"(laA + bo + 16u), "v"(a0 + 16u) : "memory");
        asm volatile("global_load_async_to_lds_b128 %0, %1, off"
                     :: "v"(laB + bo), "v"(b0) : "memory");
        asm volatile("global_load_async_to_lds_b128 %0, %1, off"
                     :: "v"(laB + bo + 16u), "v"(b0 + 16u) : "memory");
    };

    f32x8 acc[2][4] = {};
    issue(0, 0);
    int parity = 0;

    for (int k0 = 0; k0 < K; k0 += 32) {
        if (k0 + 32 < K) {
            issue(parity ^ 1, k0 + 32);
            // previous batch (4 older async ops) must have landed
            asm volatile("s_wait_asynccnt 0x4" ::: "memory");
        } else {
            asm volatile("s_wait_asynccnt 0x0" ::: "memory");
        }
        __syncthreads();

        const char* sa = smem + parity * BUF_BYTES;
        const char* sb = sa + TILE_BYTES;

        bf16x16 afrag[2], bfrag[4];
#pragma unroll
        for (int mi = 0; mi < 2; mi++) {
            // A 16x32 frag: lanes 0-15 K=0..15, lanes 16-31 K=16..31
            const char* p = sa + (wm * 32 + mi * 16 + l16) * LDS_STRIDE + lh * 32;
            bf16x8 lo = *(const bf16x8*)p;
            bf16x8 hi = *(const bf16x8*)(p + 16);
            afrag[mi] = __builtin_shufflevector(lo, hi, 0, 1, 2, 3, 4, 5, 6, 7,
                                                8, 9, 10, 11, 12, 13, 14, 15);
        }
#pragma unroll
        for (int ni = 0; ni < 4; ni++) {
            // B 32x16 frag: lane n -> 16 contiguous bf16 of weight row n
            bfrag[ni] = *(const bf16x16*)(sb + (wn * 64 + ni * 16 + l16) * LDS_STRIDE +
                                          lh * 32);
        }
#pragma unroll
        for (int mi = 0; mi < 2; mi++)
#pragma unroll
            for (int ni = 0; ni < 4; ni++)
                acc[mi][ni] = __builtin_amdgcn_wmma_f32_16x16x32_bf16(
                    false, afrag[mi], false, bfrag[ni], (short)0, acc[mi][ni],
                    false, false);
        __syncthreads();  // protect buffer being refilled next iteration
        parity ^= 1;
    }

    // epilogue: C/D layout — VGPR r, lanes 0-15: M=r, lanes 16-31: M=8+r, N=lane&15
#pragma unroll
    for (int mi = 0; mi < 2; mi++) {
#pragma unroll
        for (int ni = 0; ni < 4; ni++) {
            const int mb = bM + wm * 32 + mi * 16 + lh * 8;
            const int n = bN + wn * 64 + ni * 16 + l16;
            const float bv = bias ? bias[n] : 0.0f;
#pragma unroll
            for (int r = 0; r < 8; r++) {
                size_t idx = (size_t)(mb + r) * N + n;
                float v = acc[mi][ni][r] + bv;
                if (Res) v += Res[idx];
                C[idx] = v;
            }
        }
    }
}

// ---------------------------------------------------------------------------
// causal depthwise conv (K=4) on z = u[:, D:2D], + bias. Writes f32 and bf16.
// ---------------------------------------------------------------------------
__global__ __launch_bounds__(256) void dwconv_kernel(const float* __restrict__ u,
                                                     const float* __restrict__ cw,
                                                     const float* __restrict__ cb,
                                                     float* __restrict__ zf,
                                                     bf16* __restrict__ zb,
                                                     int T, int D) {
    const size_t i = (size_t)blockIdx.x * blockDim.x + threadIdx.x;  // over B*T*D
    const int c = (int)(i % D);
    const size_t bt = i / D;
    const int t = (int)(bt % T);
    const size_t b = bt / T;
    const size_t row2 = ((size_t)b * T) * 2 * D;
    float acc = cb[c];
#pragma unroll
    for (int k = 0; k < 4; k++) {
        int ts = t - 3 + k;
        if (ts >= 0)
            acc += cw[c * 4 + k] * u[row2 + (size_t)ts * 2 * D + D + c];
    }
    zf[i] = acc;
    zb[i] = (bf16)acc;
}

// ---------------------------------------------------------------------------
// RG-LRU scan, 3-pass chunked
// ---------------------------------------------------------------------------
__device__ __forceinline__ void lru_step(float forget, float inp, float z, float coef,
                                         float& h) {
    float a = __expf(coef * sigmoid_f(forget));
    float beta = sqrtf(fmaxf(1.0f - a * a + 1e-6f, 0.0f));
    float xs = beta * sigmoid_f(inp) * z;
    h = a * h + xs;
}

__global__ __launch_bounds__(256) void scan_pass1_kernel(
    const float* __restrict__ fg, const float* __restrict__ zf,
    const float* __restrict__ fbase, float* __restrict__ aprod,
    float* __restrict__ hlast, int T, int D, int NCH, int CL) {
    const int idx = blockIdx.x * blockDim.x + threadIdx.x;  // over B*NCH*D
    const int c = idx % D;
    const int ch = (idx / D) % NCH;
    const int b = idx / (D * NCH);
    const float coef = -8.0f * log1pf(__expf(fbase[c]));
    const int t0 = ch * CL;
    float h = 0.0f, ap = 1.0f;
    size_t o2 = ((size_t)b * T + t0) * 2 * D;
    size_t o1 = ((size_t)b * T + t0) * D;
    for (int t = 0; t < CL; t++) {
        float a = __expf(coef * sigmoid_f(fg[o2 + c]));
        float beta = sqrtf(fmaxf(1.0f - a * a + 1e-6f, 0.0f));
        float xs = beta * sigmoid_f(fg[o2 + D + c]) * zf[o1 + c];
        h = a * h + xs;
        ap *= a;
        o2 += 2 * (size_t)D;
        o1 += D;
    }
    aprod[idx] = ap;
    hlast[idx] = h;
}

__global__ __launch_bounds__(256) void scan_pass2_kernel(
    const float* __restrict__ aprod, const float* __restrict__ hlast,
    float* __restrict__ carry, int D, int NCH) {
    const int idx = blockIdx.x * blockDim.x + threadIdx.x;  // over B*D
    const int c = idx % D;
    const int b = idx / D;
    float h = 0.0f;
    for (int ch = 0; ch < NCH; ch++) {
        const size_t k = ((size_t)b * NCH + ch) * D + c;
        carry[k] = h;
        h = aprod[k] * h + hlast[k];
    }
}

__global__ __launch_bounds__(256) void scan_pass3_kernel(
    const float* __restrict__ u, const float* __restrict__ fg,
    const float* __restrict__ zf, const float* __restrict__ fbase,
    const float* __restrict__ carry, bf16* __restrict__ hg,
    int T, int D, int NCH, int CL) {
    const int idx = blockIdx.x * blockDim.x + threadIdx.x;  // over B*NCH*D
    const int c = idx % D;
    const int ch = (idx / D) % NCH;
    const int b = idx / (D * NCH);
    const float coef = -8.0f * log1pf(__expf(fbase[c]));
    const int t0 = ch * CL;
    float h = carry[idx];
    size_t o2 = ((size_t)b * T + t0) * 2 * D;
    size_t o1 = ((size_t)b * T + t0) * D;
    for (int t = 0; t < CL; t++) {
        lru_step(fg[o2 + c], fg[o2 + D + c], zf[o1 + c], coef, h);
        float g = u[o2 + c];
        hg[o1 + c] = (bf16)(gelu_f(g) * h);
        o2 += 2 * (size_t)D;
        o1 += D;
    }
}

// ---------------------------------------------------------------------------
// gated-MLP activation: v2 = gelu(ug[:, :H]) * ug[:, H:]  (bf16)
// ---------------------------------------------------------------------------
__global__ __launch_bounds__(256) void gelu_mul_kernel(const float* __restrict__ ug,
                                                       bf16* __restrict__ v2, int H) {
    const size_t i = (size_t)blockIdx.x * blockDim.x + threadIdx.x;  // over BT*H
    const size_t m = i / H;
    const int j = (int)(i % H);
    const float g = ug[m * 2 * H + j];
    const float v = ug[m * 2 * H + H + j];
    v2[i] = (bf16)(gelu_f(g) * v);
}

// ---------------------------------------------------------------------------
// launch
// ---------------------------------------------------------------------------
extern "C" void kernel_launch(void* const* d_in, const int* in_sizes, int n_in,
                              void* d_out, int out_size, void* d_ws, size_t ws_size,
                              hipStream_t stream) {
    (void)in_sizes; (void)n_in; (void)out_size; (void)ws_size;
    const int B = 4, T = 4096, D = 1024, H = 2048;
    const size_t BT = (size_t)B * T;
    const int NCH = 32, CL = T / NCH;

    const float* x        = (const float*)d_in[0];
    const float* hgamma   = (const float*)d_in[1];
    const float* input_w  = (const float*)d_in[2];
    const float* conv_w   = (const float*)d_in[3];
    const float* conv_b   = (const float*)d_in[4];
    const float* gates_w  = (const float*)d_in[5];
    const float* gates_b  = (const float*)d_in[6];
    const float* fbase    = (const float*)d_in[7];
    const float* output_w = (const float*)d_in[8];
    const float* ggamma   = (const float*)d_in[9];
    const float* grow_w   = (const float*)d_in[10];
    const float* shrink_w = (const float*)d_in[11];
    float* out = (float*)d_out;

    char* base = (char*)d_ws;
    size_t off = 0;
    auto alloc = [&](size_t bytes) -> void* {
        void* p = base + off;
        off += (bytes + 255) & ~(size_t)255;
        return p;
    };
    bf16* w_in   = (bf16*)alloc((size_t)2 * D * D * sizeof(bf16));
    bf16* w_gts  = (bf16*)alloc((size_t)2 * D * D * sizeof(bf16));
    bf16* w_out  = (bf16*)alloc((size_t)D * D * sizeof(bf16));
    bf16* w_grow = (bf16*)alloc((size_t)2 * H * D * sizeof(bf16));
    bf16* w_shr  = (bf16*)alloc((size_t)D * H * sizeof(bf16));
    bf16* xn     = (bf16*)alloc(BT * D * sizeof(bf16));       // reused as xn2
    float* u     = (float*)alloc(BT * 2 * H * sizeof(float)); // u [BT,2D], later ug [BT,2H]
    float* fg    = (float*)alloc(BT * 2 * D * sizeof(float)); // later: x2 + v2
    float* zf    = (float*)alloc(BT * D * sizeof(float));
    bf16* zb     = (bf16*)alloc(BT * D * sizeof(bf16));
    bf16* hg     = (bf16*)alloc(BT * D * sizeof(bf16));
    float* aprod = (float*)alloc((size_t)B * NCH * D * sizeof(float));
    float* hlast = (float*)alloc((size_t)B * NCH * D * sizeof(float));
    float* carry = (float*)alloc((size_t)B * NCH * D * sizeof(float));
    float* ug = u;
    float* x2 = fg;
    bf16* v2 = (bf16*)(fg + BT * D);

    const float SQRTD = 32.0f;  // sqrt(1024)

    // 1) weights -> bf16 (22 MB total, L2-resident)
    f32_to_bf16_kernel<<<2048, 256, 0, stream>>>(input_w,  w_in,   (size_t)2 * D * D);
    f32_to_bf16_kernel<<<2048, 256, 0, stream>>>(gates_w,  w_gts,  (size_t)2 * D * D);
    f32_to_bf16_kernel<<<2048, 256, 0, stream>>>(output_w, w_out,  (size_t)D * D);
    f32_to_bf16_kernel<<<2048, 256, 0, stream>>>(grow_w,   w_grow, (size_t)2 * H * D);
    f32_to_bf16_kernel<<<2048, 256, 0, stream>>>(shrink_w, w_shr,  (size_t)D * H);

    // 2) rmsnorm(x) -> bf16
    rmsnorm_bf16_kernel<<<(int)BT, 256, 0, stream>>>(x, hgamma, xn, D, SQRTD);

    // 3) u = xn @ input_w^T   [BT, 2D]
    gemm_bf16_wmma_kernel<<<dim3(2 * D / 128, (int)(BT / 128)), 256, 0, stream>>>(
        xn, w_in, nullptr, nullptr, u, (int)BT, 2 * D, D);

    // 4) causal depthwise conv on z = u[:, D:]
    dwconv_kernel<<<(int)(BT * D / 256), 256, 0, stream>>>(u, conv_w, conv_b, zf, zb, T, D);

    // 5) fg = z_conv @ gates_w^T + gates_b   [BT, 2D]
    gemm_bf16_wmma_kernel<<<dim3(2 * D / 128, (int)(BT / 128)), 256, 0, stream>>>(
        zb, w_gts, gates_b, nullptr, fg, (int)BT, 2 * D, D);

    // 6) chunked RG-LRU scan; hgate = gelu(gate)*h fused into pass 3
    scan_pass1_kernel<<<B * NCH * D / 256, 256, 0, stream>>>(fg, zf, fbase, aprod, hlast,
                                                             T, D, NCH, CL);
    scan_pass2_kernel<<<B * D / 256, 256, 0, stream>>>(aprod, hlast, carry, D, NCH);
    scan_pass3_kernel<<<B * NCH * D / 256, 256, 0, stream>>>(u, fg, zf, fbase, carry, hg,
                                                             T, D, NCH, CL);

    // 7) x2 = x + hgate @ output_w^T   [BT, D]
    gemm_bf16_wmma_kernel<<<dim3(D / 128, (int)(BT / 128)), 256, 0, stream>>>(
        hg, w_out, nullptr, x, x2, (int)BT, D, D);

    // 8) rmsnorm(x2) -> bf16
    rmsnorm_bf16_kernel<<<(int)BT, 256, 0, stream>>>(x2, ggamma, xn, D, SQRTD);

    // 9) ug = xn2 @ grow_w^T   [BT, 2H]
    gemm_bf16_wmma_kernel<<<dim3(2 * H / 128, (int)(BT / 128)), 256, 0, stream>>>(
        xn, w_grow, nullptr, nullptr, ug, (int)BT, 2 * H, D);

    // 10) v2 = gelu(gate2) * v   [BT, H] bf16
    gelu_mul_kernel<<<(int)(BT * H / 256), 256, 0, stream>>>(ug, v2, H);

    // 11) out = x2 + v2 @ shrink_w^T   [BT, D]
    gemm_bf16_wmma_kernel<<<dim3(D / 128, (int)(BT / 128)), 256, 0, stream>>>(
        v2, w_shr, nullptr, x2, out, (int)BT, D, H);
}
